// Feature_Embedder_MADDG_21861383536704
// MI455X (gfx1250) — compile-verified, matrix operand order checked by reading, not run on hardware
//
#include <hip/hip_runtime.h>
#include <hip/hip_bf16.h>

typedef __attribute__((ext_vector_type(16))) _Float16 v16h;
typedef __attribute__((ext_vector_type(8)))  _Float16 v8h;
typedef __attribute__((ext_vector_type(4)))  _Float16 v4h;
typedef __attribute__((ext_vector_type(8)))  float    v8f;
typedef __attribute__((ext_vector_type(4)))  float    v4f;
typedef int v4i __attribute__((vector_size(16)));

#define ALPHA   50.0f
#define BN_EPS  1e-5f
#define L2_EPS  1e-12f

// N=32, C=128, Hout=Wout=64 -> HW=4096, K=64 clusters, conv Kdim = 128*9 = 1152

#ifndef __has_builtin
#define __has_builtin(x) 0
#endif
#if __has_builtin(__builtin_amdgcn_global_load_async_to_lds_b128) && \
    __has_builtin(__builtin_amdgcn_s_wait_asynccnt)
#define USE_ASYNC_LDS 1
#else
#define USE_ASYNC_LDS 0
#endif

// 16-byte global -> LDS copy: async-to-LDS (ASYNCcnt path) when available.
__device__ __forceinline__ void cp16(_Float16* lds_dst, const _Float16* gsrc) {
#if USE_ASYNC_LDS
    __builtin_amdgcn_global_load_async_to_lds_b128((v4i*)gsrc, (v4i*)lds_dst, 0, 0);
#else
    *(v8h*)lds_dst = *(const v8h*)gsrc;
#endif
}
__device__ __forceinline__ void async_wait() {
#if USE_ASYNC_LDS
    __builtin_amdgcn_s_wait_asynccnt(0);
#endif
}

__device__ __forceinline__ v8f zero8() {
    v8f z;
#pragma unroll
    for (int i = 0; i < 8; ++i) z[i] = 0.f;
    return z;
}

// 16x32 f16 fragment loader (A layout; B tiles stored transposed [N][32] in
// LDS so the same loader serves both).  CDNA5 ISA 7.12.2 layout.
__device__ __forceinline__ v16h load_frag(const _Float16* base, int row0) {
    const int lane = threadIdx.x & 31;
    const int r    = row0 + (lane & 15);
    const int kb   = (lane >> 4) * 8;
    const v8h* p0 = (const v8h*)(base + r * 32 + kb);
    const v8h* p1 = (const v8h*)(base + r * 32 + kb + 16);
    v8h lo = *p0, hi = *p1;
    v16h out;
#pragma unroll
    for (int j = 0; j < 8; ++j) { out[j] = lo[j]; out[8 + j] = hi[j]; }
    return out;
}

// ---------------------------------------------------------------------------
// Kernel 0a: stream-convert input to f16 once.  The f16 image (134 MB) fits
// in the 192 MB global L2, so the conv's 9-tap im2col gather re-reads become
// L2 hits instead of repeated HBM traffic.
// ---------------------------------------------------------------------------
__global__ void convert_x(const float* __restrict__ x,
                          _Float16* __restrict__ x16) {
    const v4f* src = (const v4f*)x;
    v4h* dst = (v4h*)x16;
    size_t i = (size_t)blockIdx.x * blockDim.x + threadIdx.x;
    const size_t stride = (size_t)gridDim.x * blockDim.x;
    for (; i < 16777216ull; i += stride) {   // 67108864 floats / 4
        v4f in = src[i];
        v4h o;
#pragma unroll
        for (int j = 0; j < 4; ++j) o[j] = (_Float16)in[j];
        dst[i] = o;
    }
}

// ---------------------------------------------------------------------------
// Kernel 0b: weight prep.
// ---------------------------------------------------------------------------
__global__ void prep_kernel(const float* __restrict__ conv_w,
                            const float* __restrict__ vlad_w,
                            _Float16* __restrict__ wq,
                            _Float16* __restrict__ wnT,
                            float* __restrict__ asum) {
    const int t = blockIdx.x * blockDim.x + threadIdx.x;
    const int stride = gridDim.x * blockDim.x;
    for (int i = t; i < 128 * 1152; i += stride) wq[i] = (_Float16)conv_w[i];
    for (int i = t; i < 32 * 64; i += stride) asum[i] = 0.f;
    if (t < 64) {
        float s = 0.f;
        for (int c = 0; c < 128; ++c) { float v = vlad_w[c * 64 + t]; s += v * v; }
        float sc = 1.f / fmaxf(sqrtf(s), L2_EPS);
        for (int c = 0; c < 128; ++c)
            wnT[t * 128 + c] = (_Float16)(vlad_w[c * 64 + t] * sc);
    }
}

// ---------------------------------------------------------------------------
// Kernel 1: implicit-GEMM conv3x3 s2 p1 + BN + ReLU + channel L2 norm.
// Block = 128 Cout x 64 positions; grid = 2048; 8 waves.
// Double-buffered LDS tiles; f16 L2-resident input; (cin,ky,kx) LUT decode.
// ---------------------------------------------------------------------------
__global__ void conv_bn_relu_norm(const _Float16* __restrict__ x16,
                                  const _Float16* __restrict__ wq,
                                  const float* __restrict__ gamma,
                                  const float* __restrict__ beta,
                                  const float* __restrict__ mean,
                                  const float* __restrict__ var,
                                  float* __restrict__ featOut,    // (N,HW,C) f32
                                  _Float16* __restrict__ xn,      // (N,HW,C) f16
                                  _Float16* __restrict__ xnT) {   // (N,C,HW) f16
    __shared__ __align__(16) _Float16 As[2][128 * 32];
    __shared__ __align__(16) _Float16 Bt[2][64 * 32];
    __shared__ int   ktab[1152];
    __shared__ float inv_s[128], b2_s[128];
    __shared__ float possum[64];

    const int tid  = threadIdx.x;
    const int lane = tid & 31;
    const int wv   = tid >> 5;
    const int n    = blockIdx.x >> 6;
    const int ibase = (blockIdx.x & 63) * 64;

    if (tid < 128) {
        float iv = gamma[tid] * rsqrtf(var[tid] + BN_EPS);
        inv_s[tid] = iv;
        b2_s[tid]  = beta[tid] - mean[tid] * iv;
    }
    if (tid < 64) possum[tid] = 0.f;
    for (int i = tid; i < 1152; i += 256) {
        int cin = i / 9, r9 = i - cin * 9;
        int ky = r9 / 3, kx = r9 - ky * 3;
        ktab[i] = (cin << 4) | (ky << 2) | kx;
    }
    __syncthreads();

    // per-thread constants for the tile loaders (invariant across chunks)
    const int aE  = tid * 16, aCo = aE >> 5, aT0 = aE & 31;      // A tile
    const int bE  = tid * 8,  bP  = bE >> 5, bT0 = bE & 31;      // B tile
    const int oy  = (ibase + bP) >> 6, ox = (ibase + bP) & 63;
    const int iy0 = oy * 2 - 1, ix0 = ox * 2 - 1;
    const _Float16* xin = x16 + (size_t)n * 128 * 128 * 128;

    auto load_chunk = [&](int kc, int buf) {
        // A: weights 128x32, contiguous 32B per thread (async to LDS)
        const _Float16* asrc = wq + aCo * 1152 + kc * 32 + aT0;
        cp16(&As[buf][aCo * 32 + aT0], asrc);
        cp16(&As[buf][aCo * 32 + aT0 + 8], asrc + 8);
        // B: im2col gather (f16, L2-resident), Bt[pos][k]
#pragma unroll
        for (int j = 0; j < 8; ++j) {
            int kt_ = ktab[kc * 32 + bT0 + j];
            int cin = kt_ >> 4, ky = (kt_ >> 2) & 3, kx = kt_ & 3;
            int iy = iy0 + ky, ix = ix0 + kx;
            _Float16 v = (_Float16)0.f;
            if ((unsigned)iy < 128u && (unsigned)ix < 128u)
                v = xin[((size_t)cin * 128 + iy) * 128 + ix];
            Bt[buf][bP * 32 + bT0 + j] = v;
        }
    };

    v8f acc[4];
#pragma unroll
    for (int j = 0; j < 4; ++j) acc[j] = zero8();

    load_chunk(0, 0);
    for (int kc = 0; kc < 36; ++kc) {
        const int cur = kc & 1;
        async_wait();
        __syncthreads();
        if (kc + 1 < 36) load_chunk(kc + 1, cur ^ 1);
        v16h af = load_frag(As[cur], wv * 16);
#pragma unroll
        for (int pt = 0; pt < 4; ++pt) {
            v16h bf = load_frag(Bt[cur], pt * 16);
            acc[pt] = __builtin_amdgcn_wmma_f32_16x16x32_f16(
                false, af, false, bf, (short)0, acc[pt], false, false);
        }
    }
    __syncthreads();

    // Epilogue: BN + ReLU + per-position squared-sum reduction
    const int hi   = lane >> 4;
    const int pcol = lane & 15;
#pragma unroll
    for (int pt = 0; pt < 4; ++pt) {
        int p = pt * 16 + pcol;
        float ss = 0.f;
#pragma unroll
        for (int r = 0; r < 8; ++r) {
            int c = wv * 16 + 8 * hi + r;
            float v = acc[pt][r] * inv_s[c] + b2_s[c];
            v = fmaxf(v, 0.f);
            acc[pt][r] = v;
            ss += v * v;
        }
        atomicAdd(&possum[p], ss);
    }
    __syncthreads();

    const size_t fbase = ((size_t)n * 4096 + ibase) * 128;
#pragma unroll
    for (int pt = 0; pt < 4; ++pt) {
        int p = pt * 16 + pcol;
        float scale = 1.f / fmaxf(sqrtf(possum[p]), L2_EPS);
        int c0 = wv * 16 + 8 * hi;
        size_t o = fbase + (size_t)p * 128 + c0;
#pragma unroll
        for (int r = 0; r < 8; ++r) {
            float v = acc[pt][r] * scale;
            featOut[o + r] = v;
            xn[o + r] = (_Float16)v;
            xnT[((size_t)n * 128 + c0 + r) * 4096 + ibase + p] = (_Float16)v;
        }
    }
}

// ---------------------------------------------------------------------------
// Kernel 2: soft-assign GEMM + bias + softmax over K.
// Block = 64 positions x 64 clusters; grid = 2048; 8 waves.
// ---------------------------------------------------------------------------
__global__ void assign_softmax(const _Float16* __restrict__ xn,   // (N,HW,C)
                               const _Float16* __restrict__ wnT,  // (K,C)
                               const float* __restrict__ vb,      // (K)
                               float* __restrict__ saOut,         // (N,K,HW)
                               _Float16* __restrict__ a_ws,       // (N,K,HW)
                               float* __restrict__ asum) {        // (N,K)
    __shared__ __align__(16) _Float16 At[64 * 32];
    __shared__ __align__(16) _Float16 Bt[64 * 32];
    __shared__ float Lg[64 * 65];
    __shared__ float bias_s[64];

    const int tid  = threadIdx.x;
    const int lane = tid & 31;
    const int wv   = tid >> 5;
    const int n    = blockIdx.x >> 6;
    const int ibase = (blockIdx.x & 63) * 64;

    if (tid < 64) bias_s[tid] = vb[tid];

    const int e = tid * 8, row = e >> 5, t0 = e & 31;
    const int pt = wv & 3;
    const int ctb = (wv >> 2) * 2;
    v8f acc[2];
    acc[0] = zero8(); acc[1] = zero8();

    for (int cc = 0; cc < 4; ++cc) {
        __syncthreads();
        cp16(&At[row * 32 + t0],
             xn + ((size_t)n * 4096 + ibase + row) * 128 + cc * 32 + t0);
        cp16(&Bt[row * 32 + t0], wnT + row * 128 + cc * 32 + t0);
        async_wait();
        __syncthreads();
        v16h af = load_frag(At, pt * 16);
#pragma unroll
        for (int j = 0; j < 2; ++j) {
            v16h bf = load_frag(Bt, (ctb + j) * 16);
            acc[j] = __builtin_amdgcn_wmma_f32_16x16x32_f16(
                false, af, false, bf, (short)0, acc[j], false, false);
        }
    }

    const int hi = lane >> 4;
#pragma unroll
    for (int j = 0; j < 2; ++j) {
        int ncl = (ctb + j) * 16 + (lane & 15);
#pragma unroll
        for (int r = 0; r < 8; ++r) {
            int p = pt * 16 + 8 * hi + r;
            float v = acc[j][r];
            saOut[((size_t)n * 64 + ncl) * 4096 + ibase + p] = v;   // pre-bias
            Lg[p * 65 + ncl] = ALPHA * (v + bias_s[ncl]);
        }
    }
    __syncthreads();

    if (tid < 64) {
        float* rowp = &Lg[tid * 65];
        float m = -3.4e38f;
        for (int k = 0; k < 64; ++k) m = fmaxf(m, rowp[k]);
        float s = 0.f;
        for (int k = 0; k < 64; ++k) { float ee = __expf(rowp[k] - m); rowp[k] = ee; s += ee; }
        float inv = 1.f / s;
        for (int k = 0; k < 64; ++k) rowp[k] *= inv;
    }
    __syncthreads();

    for (int idx = tid; idx < 4096; idx += 256) {
        int k = idx >> 6, p = idx & 63;
        a_ws[((size_t)n * 64 + k) * 4096 + ibase + p] = (_Float16)Lg[p * 65 + k];
    }
    if (tid < 64) {
        float s = 0.f;
        for (int p = 0; p < 64; ++p) s += Lg[p * 65 + tid];
        atomicAdd(&asum[n * 64 + tid], s);
    }
}

// ---------------------------------------------------------------------------
// Kernel 3a: split-K VLAD GEMM.  grid = N*8 blocks; block reduces 512 HW
// positions into a raw f32 partial (N,8,K,C).  Double-buffered tiles.
// ---------------------------------------------------------------------------
__global__ void vlad_partial(const _Float16* __restrict__ a_ws,   // (N,K,HW)
                             const _Float16* __restrict__ xnT,    // (N,C,HW)
                             float* __restrict__ part) {          // (N,8,K,C)
    __shared__ __align__(16) _Float16 At[2][64 * 32];
    __shared__ __align__(16) _Float16 Bt[2][128 * 32];

    const int tid  = threadIdx.x;
    const int lane = tid & 31;
    const int wv   = tid >> 5;
    const int n    = blockIdx.x >> 3;
    const int hw0  = (blockIdx.x & 7) * 512;

    const int aE = tid * 8,  aK = aE >> 5, aT0 = aE & 31;
    const int bE = tid * 16, bC = bE >> 5, bT0 = bE & 31;

    auto load_chunk = [&](int kc, int buf) {
        int hwb = hw0 + kc * 32;
        cp16(&At[buf][aK * 32 + aT0],
             a_ws + ((size_t)n * 64 + aK) * 4096 + hwb + aT0);
        const _Float16* bs = xnT + ((size_t)n * 128 + bC) * 4096 + hwb + bT0;
        cp16(&Bt[buf][bC * 32 + bT0], bs);
        cp16(&Bt[buf][bC * 32 + bT0 + 8], bs + 8);
    };

    const int kt  = wv & 3;
    const int ctb = (wv >> 2) * 4;
    v8f acc[4];
#pragma unroll
    for (int j = 0; j < 4; ++j) acc[j] = zero8();

    load_chunk(0, 0);
    for (int kc = 0; kc < 16; ++kc) {
        const int cur = kc & 1;
        async_wait();
        __syncthreads();
        if (kc + 1 < 16) load_chunk(kc + 1, cur ^ 1);
        v16h af = load_frag(At[cur], kt * 16);
#pragma unroll
        for (int j = 0; j < 4; ++j) {
            v16h bf = load_frag(Bt[cur], (ctb + j) * 16);
            acc[j] = __builtin_amdgcn_wmma_f32_16x16x32_f16(
                false, af, false, bf, (short)0, acc[j], false, false);
        }
    }

    const int hi = lane >> 4;
    float* pb = part + (size_t)blockIdx.x * 8192;   // (n*8+seg)*64*128
#pragma unroll
    for (int r = 0; r < 8; ++r) {
        int k = kt * 16 + 8 * hi + r;
#pragma unroll
        for (int j = 0; j < 4; ++j) {
            int c = (ctb + j) * 16 + (lane & 15);
            pb[k * 128 + c] = acc[j][r];
        }
    }
}

// ---------------------------------------------------------------------------
// Kernel 3b: reduce partials, subtract asum*c_k, intra-norm + global L2.
// One block per image; thread owns one quarter-row (k = tid>>2, 32 channels).
// ---------------------------------------------------------------------------
__global__ void vlad_finalize(const float* __restrict__ part,    // (N,8,K,C)
                              const float* __restrict__ asum,    // (N,K)
                              const float* __restrict__ vlad_w,  // (C,K)
                              float* __restrict__ vladOut) {     // (N,K*C)
    __shared__ float rowsq[64], rowscale[64], contrib[64];
    __shared__ float gscale_s;

    const int tid = threadIdx.x;
    const int n   = blockIdx.x;
    const int k   = tid >> 2;
    const int c0  = (tid & 3) * 32;

    if (tid < 64) rowsq[tid] = 0.f;
    __syncthreads();

    const float* pb = part + (size_t)n * 8 * 8192 + k * 128 + c0;
    const float ak  = asum[n * 64 + k];
    float vals[32];
    float ss = 0.f;
#pragma unroll 4
    for (int ii = 0; ii < 32; ++ii) {
        float s = 0.f;
#pragma unroll
        for (int sg = 0; sg < 8; ++sg) s += pb[sg * 8192 + ii];
        s -= ak * vlad_w[(c0 + ii) * 64 + k];
        vals[ii] = s;
        ss += s * s;
    }
    atomicAdd(&rowsq[k], ss);
    __syncthreads();

    if (tid < 64) {
        float rq = rowsq[tid];
        float sc = 1.f / fmaxf(sqrtf(rq), L2_EPS);
        rowscale[tid] = sc;
        contrib[tid]  = rq * sc * sc;
    }
    __syncthreads();
    if (tid == 0) {
        float tot = 0.f;
        for (int kk = 0; kk < 64; ++kk) tot += contrib[kk];
        gscale_s = 1.f / fmaxf(sqrtf(tot), L2_EPS);
    }
    __syncthreads();

    const float sc = rowscale[k] * gscale_s;
    float* ob = vladOut + (size_t)n * 8192 + k * 128 + c0;
#pragma unroll 4
    for (int ii = 0; ii < 32; ++ii) ob[ii] = vals[ii] * sc;
}

// ---------------------------------------------------------------------------
extern "C" void kernel_launch(void* const* d_in, const int* in_sizes, int n_in,
                              void* d_out, int out_size, void* d_ws, size_t ws_size,
                              hipStream_t stream) {
    (void)in_sizes; (void)n_in; (void)out_size; (void)ws_size;
    const float* x      = (const float*)d_in[0];
    const float* conv_w = (const float*)d_in[1];
    const float* gamma  = (const float*)d_in[2];
    const float* beta   = (const float*)d_in[3];
    const float* mean   = (const float*)d_in[4];
    const float* var    = (const float*)d_in[5];
    const float* vlad_w = (const float*)d_in[6];
    const float* vlad_b = (const float*)d_in[7];
    // d_in[8] = norm_flag (no effect on reference math path)

    float* out = (float*)d_out;
    float* vladOut = out;                 // 32*8192
    float* saOut   = out + 262144;        // 32*64*4096
    float* featOut = out + 8650752;       // 32*4096*128

    char* w = (char*)d_ws;
    _Float16* wq   = (_Float16*)(w);                          // 128*1152 f16
    _Float16* wnT  = (_Float16*)(w + 294912);                 // 64*128   f16
    float*    asum = (float*)   (w + 311296);                 // 32*64    f32
    _Float16* xn   = (_Float16*)(w + 319488);                 // (N,HW,C) f16
    _Float16* xnT  = (_Float16*)(w + 319488 + 33554432ull);   // (N,C,HW) f16
    _Float16* a_ws = (_Float16*)(w + 319488 + 67108864ull);   // (N,K,HW) f16
    float*    part = (float*)   (w + 84205568ull);            // (N,8,K,C) f32
    _Float16* x16  = (_Float16*)(w + 92594176ull);            // (N,C,H,W) f16

    convert_x<<<2048, 256, 0, stream>>>(x, x16);
    prep_kernel<<<72, 256, 0, stream>>>(conv_w, vlad_w, wq, wnT, asum);
    conv_bn_relu_norm<<<2048, 256, 0, stream>>>(x16, wq, gamma, beta, mean, var,
                                                featOut, xn, xnT);
    assign_softmax<<<2048, 256, 0, stream>>>(xn, wnT, vlad_b, saOut, a_ws, asum);
    vlad_partial<<<256, 256, 0, stream>>>(a_ws, xnT, part);
    vlad_finalize<<<32, 256, 0, stream>>>(part, asum, vlad_w, vladOut);
}
